// Block_66546223284681
// MI455X (gfx1250) — matrix operand order repaired and checked by reading, not compiled, not run on hardware
//
#include <hip/hip_runtime.h>
#include <hip/hip_bf16.h>
#include <math.h>

typedef _Float16 f16;
typedef _Float16 v16h __attribute__((ext_vector_type(16)));
typedef _Float16 v8h  __attribute__((ext_vector_type(8)));
typedef float    v8f  __attribute__((ext_vector_type(8)));
typedef int      v4i  __attribute__((vector_size(4 * sizeof(int))));

// ---- probe the gfx1250 async global->LDS path; fall back to sync copies ----
#if defined(__gfx1250__) && \
    __has_builtin(__builtin_amdgcn_global_load_async_to_lds_b128) && \
    __has_builtin(__builtin_amdgcn_s_wait_asynccnt)
#define USE_ASYNC_LDS 1
#else
#define USE_ASYNC_LDS 0
#endif

static __device__ __forceinline__ void copy16_g2lds(const f16* g, f16* l) {
#if USE_ASYNC_LDS
  __builtin_amdgcn_global_load_async_to_lds_b128(
      (__attribute__((address_space(1))) v4i*)g,
      (__attribute__((address_space(3))) v4i*)l, 0, 0);
#else
  *(v8h*)l = *(const v8h*)g;
#endif
}

static __device__ __forceinline__ void async_fence() {
#if USE_ASYNC_LDS
  __builtin_amdgcn_s_wait_asynccnt(0);
#endif
}

static __device__ __forceinline__ v8f vzero() {
  v8f z;
#pragma unroll
  for (int i = 0; i < 8; ++i) z[i] = 0.f;
  return z;
}

static __device__ __forceinline__ v8f wmma32(v16h a, v16h b, v8f c) {
  return __builtin_amdgcn_wmma_f32_16x16x32_f16(false, a, false, b, (short)0, c,
                                                false, false);
}

// A-fragment (16x32 f16): lane L holds row L&15; K cols kh..kh+15
// with kh = 16*(L>=16).  Two 16-byte loads.
static __device__ __forceinline__ v16h load_a_frag(const f16* base, int ld) {
  const int lane = threadIdx.x & 31;
  const int row = lane & 15;
  const int kh = (lane >> 4) * 16;
  const f16* p = base + (long)row * ld + kh;
  v8h lo = *(const v8h*)(p);
  v8h hi = *(const v8h*)(p + 8);
  v16h a;
#pragma unroll
  for (int i = 0; i < 8; ++i) { a[i] = lo[i]; a[8 + i] = hi[i]; }
  return a;
}

// B-fragment (32x16 f16) loaded from a TRANSPOSED tile BT[col][k] (contraction
// contiguous): lane L = column L&15, K-rows kg..kg+15 with kg = 16*(L>=16).
// One aligned 32-byte v16h load.
static __device__ __forceinline__ v16h load_b_fragT(const f16* bt, int ld) {
  const int lane = threadIdx.x & 31;
  const int col = lane & 15;
  const int kg = (lane >> 4) * 16;
  return *(const v16h*)(bt + (long)col * ld + kg);
}

// ---------------------------------------------------------------------------
// elementwise casts
// ---------------------------------------------------------------------------
__global__ void cast_f16_kernel(const float* __restrict__ in, f16* __restrict__ out, long n) {
  long i = (long)blockIdx.x * 256 + threadIdx.x;
  if (i < n) out[i] = (f16)in[i];
}

// in: (Nn x Kk) row-major f32  ->  out: (Kk x Nn) row-major f16
__global__ void transpose_cast_kernel(const float* __restrict__ in, f16* __restrict__ out,
                                      int Nn, int Kk) {
  long i = (long)blockIdx.x * 256 + threadIdx.x;
  if (i < (long)Nn * Kk) {
    int kk = (int)(i / Nn);
    int nn = (int)(i % Nn);
    out[i] = (f16)in[(long)nn * Kk + kk];
  }
}

// ---------------------------------------------------------------------------
// LayerNorm (C = 768) -> f16
// ---------------------------------------------------------------------------
__global__ __launch_bounds__(256) void ln_cast_kernel(const float* __restrict__ x,
                                                      const float* __restrict__ w,
                                                      const float* __restrict__ bb,
                                                      f16* __restrict__ out) {
  __shared__ float red[256];
  __shared__ float stats[2];
  const int tid = threadIdx.x;
  const long base = (long)blockIdx.x * 768;
  float xv[3];
  float s = 0.f, s2 = 0.f;
#pragma unroll
  for (int i = 0; i < 3; ++i) {
    xv[i] = x[base + tid + i * 256];
    s += xv[i];
    s2 += xv[i] * xv[i];
  }
  red[tid] = s;
  __syncthreads();
  for (int o = 128; o > 0; o >>= 1) {
    if (tid < o) red[tid] += red[tid + o];
    __syncthreads();
  }
  if (tid == 0) stats[0] = red[0];
  __syncthreads();
  red[tid] = s2;
  __syncthreads();
  for (int o = 128; o > 0; o >>= 1) {
    if (tid < o) red[tid] += red[tid + o];
    __syncthreads();
  }
  if (tid == 0) stats[1] = red[0];
  __syncthreads();
  const float mu = stats[0] * (1.f / 768.f);
  const float var = stats[1] * (1.f / 768.f) - mu * mu;
  const float inv = rsqrtf(var + 1e-6f);
#pragma unroll
  for (int i = 0; i < 3; ++i) {
    int c = tid + i * 256;
    out[base + c] = (f16)((xv[i] - mu) * inv * w[c] + bb[c]);
  }
}

// ---------------------------------------------------------------------------
// Generic f16 GEMM: C(MxN) = A(MxK) * B(KxN), 128x128 block, 8 waves, BK=32.
// EPI 0: outH = f16(acc + bias)
// EPI 1: outH = f16(gelu(acc + bias))         (exact erf GELU)
// EPI 2: outF = acc + bias + resid            (f32)
// M, N multiples of 128; Kd multiple of 32.
// ---------------------------------------------------------------------------
template <int EPI>
__global__ __launch_bounds__(256) void gemm_kernel(const f16* __restrict__ A, int lda,
                                                   const f16* __restrict__ Bw, int ldb,
                                                   const float* __restrict__ bias,
                                                   const float* __restrict__ resid,
                                                   f16* __restrict__ outH,
                                                   float* __restrict__ outF, int ldc,
                                                   int Kd) {
  __shared__ __align__(32) f16 As[128 * 32];
  __shared__ __align__(32) f16 Bt[128 * 32];  // BT[n][k]
  const int tid = threadIdx.x;
  const int lane = tid & 31;
  const int wid = tid >> 5;
  const int wr = wid & 3;   // 4 wave-rows * 32
  const int wc = wid >> 2;  // 2 wave-cols * 64
  const long m0 = (long)blockIdx.y * 128;
  const long n0 = (long)blockIdx.x * 128;

  v8f acc[2][4];
#pragma unroll
  for (int i = 0; i < 2; ++i)
#pragma unroll
    for (int j = 0; j < 4; ++j) acc[i][j] = vzero();

  for (int k0 = 0; k0 < Kd; k0 += 32) {
#pragma unroll
    for (int it = 0; it < 2; ++it) {  // A tile 128x32 (async direct-to-LDS)
      int c = tid + it * 256;
      int r = c >> 2, cc = (c & 3) * 8;
      copy16_g2lds(A + (m0 + r) * lda + k0 + cc, As + r * 32 + cc);
    }
#pragma unroll
    for (int it = 0; it < 2; ++it) {  // B tile 32x128 -> transposed into LDS
      int c = tid + it * 256;
      int k = c >> 4, nc = (c & 15) * 8;
      v8h v = *(const v8h*)(Bw + (long)(k0 + k) * ldb + n0 + nc);
#pragma unroll
      for (int j = 0; j < 8; ++j) Bt[(nc + j) * 32 + k] = v[j];
    }
    if (k0 + 32 < Kd)
      __builtin_prefetch(A + (m0 + (tid >> 1)) * lda + k0 + 32, 0, 1);
    async_fence();
    __syncthreads();

    v16h af[2], bf[4];
#pragma unroll
    for (int i = 0; i < 2; ++i) af[i] = load_a_frag(As + (wr * 32 + 16 * i) * 32, 32);
#pragma unroll
    for (int j = 0; j < 4; ++j) bf[j] = load_b_fragT(Bt + (wc * 64 + 16 * j) * 32, 32);
#pragma unroll
    for (int i = 0; i < 2; ++i)
#pragma unroll
      for (int j = 0; j < 4; ++j) acc[i][j] = wmma32(af[i], bf[j], acc[i][j]);
    __syncthreads();
  }

  const int cc = lane & 15, g = lane >> 4;
#pragma unroll
  for (int i = 0; i < 2; ++i) {
#pragma unroll
    for (int j = 0; j < 4; ++j) {
      const long col = n0 + wc * 64 + 16 * j + cc;
      const float bcol = bias[col];
#pragma unroll
      for (int v = 0; v < 8; ++v) {
        const long r = m0 + wr * 32 + 16 * i + v + 8 * g;
        float val = acc[i][j][v] + bcol;
        if (EPI == 0) {
          outH[r * ldc + col] = (f16)val;
        } else if (EPI == 1) {
          float ge = 0.5f * val * (1.f + erff(val * 0.70710678118654752f));
          outH[r * ldc + col] = (f16)ge;
        } else {
          outF[r * ldc + col] = val + resid[r * ldc + col];
        }
      }
    }
  }
}

// ---------------------------------------------------------------------------
// Linformer K/V projection: out(256x64) = EprojT(256x8192) * X[b,h](8192x64)
// X is a strided view of qkv16.  STORE_T: store transposed (64x256) for v_proj.
// ---------------------------------------------------------------------------
template <bool STORE_T>
__global__ __launch_bounds__(256) void gemm_kv_kernel(const f16* __restrict__ A,
                                                      const f16* __restrict__ qkv16,
                                                      int sel, f16* __restrict__ outp) {
  __shared__ __align__(32) f16 As[128 * 32];
  __shared__ __align__(32) f16 Bt[64 * 32];
  const int tid = threadIdx.x;
  const int lane = tid & 31;
  const int wid = tid >> 5;
  const int wr = wid & 3;   // 4 * 32 rows
  const int wc = wid >> 2;  // 2 * 32 cols
  const int m0 = blockIdx.x * 128;
  const int bh = blockIdx.y;
  const int b = bh / 12, h = bh % 12;
  const f16* Bg = qkv16 + (long)b * 8192 * 2304 + sel * 768 + h * 64;

  v8f acc[2][2];
#pragma unroll
  for (int i = 0; i < 2; ++i)
#pragma unroll
    for (int j = 0; j < 2; ++j) acc[i][j] = vzero();

  for (int k0 = 0; k0 < 8192; k0 += 32) {
#pragma unroll
    for (int it = 0; it < 2; ++it) {  // A tile 128x32 (async direct-to-LDS)
      int c = tid + it * 256;
      int r = c >> 2, cc = (c & 3) * 8;
      copy16_g2lds(A + (long)(m0 + r) * 8192 + k0 + cc, As + r * 32 + cc);
    }
    {  // B tile 32x64 strided from qkv -> transposed
      int k = tid >> 3, nc = (tid & 7) * 8;
      v8h v = *(const v8h*)(Bg + (long)(k0 + k) * 2304 + nc);
#pragma unroll
      for (int j = 0; j < 8; ++j) Bt[(nc + j) * 32 + k] = v[j];
    }
    async_fence();
    __syncthreads();
    v16h af[2], bf[2];
#pragma unroll
    for (int i = 0; i < 2; ++i) af[i] = load_a_frag(As + (wr * 32 + 16 * i) * 32, 32);
#pragma unroll
    for (int j = 0; j < 2; ++j) bf[j] = load_b_fragT(Bt + (wc * 32 + 16 * j) * 32, 32);
#pragma unroll
    for (int i = 0; i < 2; ++i)
#pragma unroll
      for (int j = 0; j < 2; ++j) acc[i][j] = wmma32(af[i], bf[j], acc[i][j]);
    __syncthreads();
  }

  const int cc = lane & 15, g = lane >> 4;
  f16* ob = outp + (long)bh * 256 * 64;
#pragma unroll
  for (int i = 0; i < 2; ++i)
#pragma unroll
    for (int j = 0; j < 2; ++j) {
      const int col = wc * 32 + 16 * j + cc;
#pragma unroll
      for (int v = 0; v < 8; ++v) {
        const int r = m0 + wr * 32 + 16 * i + v + 8 * g;
        f16 hv = (f16)acc[i][j][v];
        if (STORE_T) ob[(long)col * 256 + r] = hv;  // (64x256)
        else         ob[(long)r * 64 + col] = hv;   // (256x64)
      }
    }
}

// ---------------------------------------------------------------------------
// Fused Linformer attention for one (b, h, 128-row block of N).
//   S = q_blk(128x64) * k_proj^T  -> softmax -> O = P * v_proj -> o16 (B,N,C)
// kp16:  per-bh (256x64)  (= BT of k_proj^T -> contiguous B-frags)
// vpT16: per-bh (64x256)  (= BT of v_proj   -> contiguous B-frags)
// Each of 8 waves owns 16 rows; all K=256 logits live in its accumulators.
// ---------------------------------------------------------------------------
__global__ __launch_bounds__(256) void attn_kernel(const f16* __restrict__ qkv16,
                                                   const f16* __restrict__ kp16,
                                                   const f16* __restrict__ vpT16,
                                                   f16* __restrict__ o16) {
  extern __shared__ f16 smem_h[];
  f16* kp = smem_h;                 // 256*64
  f16* vp = smem_h + 256 * 64;      // 64*256
  f16* P  = smem_h + 2 * 256 * 64;  // 8 waves * 16*256
  const int tid = threadIdx.x;
  const int lane = tid & 31;
  const int wid = tid >> 5;
  const int h = blockIdx.y, b = blockIdx.z;
  const int bh = b * 12 + h;
  {
    const f16* g1 = kp16 + (long)bh * 16384;
    const f16* g2 = vpT16 + (long)bh * 16384;
#pragma unroll
    for (int it = 0; it < 8; ++it) {
      int idx = (tid + it * 256) * 8;
      copy16_g2lds(g1 + idx, kp + idx);
      copy16_g2lds(g2 + idx, vp + idx);
    }
  }
  async_fence();
  __syncthreads();

  const int n0 = blockIdx.x * 128 + wid * 16;
  const int cc = lane & 15, g = lane >> 4;
  const int kh = g * 16, kg = g * 16;

  // q A-fragments straight from global (lda = 2304)
  v16h aq[2];
  {
    const f16* qbase = qkv16 + ((long)(b * 8192 + n0 + cc)) * 2304 + h * 64 + kh;
#pragma unroll
    for (int s = 0; s < 2; ++s) {
      v8h lo = *(const v8h*)(qbase + s * 32);
      v8h hi = *(const v8h*)(qbase + s * 32 + 8);
#pragma unroll
      for (int i = 0; i < 8; ++i) { aq[s][i] = lo[i]; aq[s][8 + i] = hi[i]; }
    }
  }

  // S = q * k_proj^T : 16 col-tiles x 2 k-steps = 32 WMMAs
  v8f sacc[16];
#pragma unroll
  for (int t = 0; t < 16; ++t) sacc[t] = vzero();
#pragma unroll
  for (int t = 0; t < 16; ++t)
#pragma unroll
    for (int s = 0; s < 2; ++s) {
      v16h bf = *(const v16h*)(kp + (t * 16 + cc) * 64 + s * 32 + kg);
      sacc[t] = wmma32(aq[s], bf, sacc[t]);
    }

  // scale, clip, rowwise softmax (rows live in 16-lane halves -> shfl_xor<16)
  float mrow[8], srow[8];
#pragma unroll
  for (int v = 0; v < 8; ++v) mrow[v] = -3.0e38f;
#pragma unroll
  for (int t = 0; t < 16; ++t)
#pragma unroll
    for (int v = 0; v < 8; ++v) {
      float xv = sacc[t][v] * 0.125f;
      xv = fminf(fmaxf(xv, -50.f), 50.f);
      sacc[t][v] = xv;
      mrow[v] = fmaxf(mrow[v], xv);
    }
#pragma unroll
  for (int mask = 1; mask < 16; mask <<= 1)
#pragma unroll
    for (int v = 0; v < 8; ++v) mrow[v] = fmaxf(mrow[v], __shfl_xor(mrow[v], mask, 32));
#pragma unroll
  for (int v = 0; v < 8; ++v) srow[v] = 0.f;
#pragma unroll
  for (int t = 0; t < 16; ++t)
#pragma unroll
    for (int v = 0; v < 8; ++v) {
      float e = __expf(sacc[t][v] - mrow[v]);
      sacc[t][v] = e;
      srow[v] += e;
    }
#pragma unroll
  for (int mask = 1; mask < 16; mask <<= 1)
#pragma unroll
    for (int v = 0; v < 8; ++v) srow[v] += __shfl_xor(srow[v], mask, 32);
#pragma unroll
  for (int v = 0; v < 8; ++v) srow[v] = 1.f / srow[v];

  // transpose P (C-layout -> A-layout) through this wave's LDS slab
  f16* Pw = P + wid * 16 * 256;
#pragma unroll
  for (int t = 0; t < 16; ++t)
#pragma unroll
    for (int v = 0; v < 8; ++v)
      Pw[(v + 8 * g) * 256 + t * 16 + cc] = (f16)(sacc[t][v] * srow[v]);
  __syncthreads();

  // O = P(16x256) * v_proj(256x64): 8 k-steps x 4 col-tiles = 32 WMMAs
  v8f oacc[4];
#pragma unroll
  for (int td = 0; td < 4; ++td) oacc[td] = vzero();
#pragma unroll
  for (int s = 0; s < 8; ++s) {
    v16h ap = load_a_frag(Pw + s * 32, 256);
#pragma unroll
    for (int td = 0; td < 4; ++td) {
      v16h bf = *(const v16h*)(vp + (td * 16 + cc) * 256 + s * 32 + kg);
      oacc[td] = wmma32(ap, bf, oacc[td]);
    }
  }
  f16* ob = o16 + ((long)(b * 8192 + n0)) * 768 + h * 64;
#pragma unroll
  for (int td = 0; td < 4; ++td)
#pragma unroll
    for (int v = 0; v < 8; ++v)
      ob[(long)(v + 8 * g) * 768 + td * 16 + cc] = (f16)oacc[td][v];
}

// ---------------------------------------------------------------------------
extern "C" void kernel_launch(void* const* d_in, const int* in_sizes, int n_in,
                              void* d_out, int out_size, void* d_ws, size_t ws_size,
                              hipStream_t stream) {
  (void)in_sizes; (void)n_in; (void)out_size; (void)ws_size;
  const float* x      = (const float*)d_in[0];
  const float* ln1_w  = (const float*)d_in[1];
  const float* ln1_b  = (const float*)d_in[2];
  const float* qkv_w  = (const float*)d_in[3];
  const float* qkv_b  = (const float*)d_in[4];
  const float* Ek     = (const float*)d_in[5];
  const float* Ev     = (const float*)d_in[6];
  const float* proj_w = (const float*)d_in[7];
  const float* proj_b = (const float*)d_in[8];
  const float* ln2_w  = (const float*)d_in[9];
  const float* ln2_b  = (const float*)d_in[10];
  const float* fc1_w  = (const float*)d_in[11];
  const float* fc1_b  = (const float*)d_in[12];
  const float* fc2_w  = (const float*)d_in[13];
  const float* fc2_b  = (const float*)d_in[14];

  char* ws = (char*)d_ws;
  size_t off = 0;
  auto alloc = [&](size_t bytes) -> char* {
    off = (off + 255) & ~(size_t)255;
    char* p = ws + off;
    off += bytes;
    return p;
  };
  f16*   qkvW16 = (f16*)alloc(768L * 2304 * 2);
  f16*   projW16= (f16*)alloc(768L * 768 * 2);
  f16*   fc1W16 = (f16*)alloc(768L * 3072 * 2);
  f16*   fc2W16 = (f16*)alloc(3072L * 768 * 2);
  f16*   EkT16  = (f16*)alloc(256L * 8192 * 2);
  f16*   EvT16  = (f16*)alloc(256L * 8192 * 2);
  f16*   hbuf16 = (f16*)alloc(16384L * 768 * 2);   // h1 -> o -> h2 (reused)
  f16*   kp16   = (f16*)alloc(24L * 256 * 64 * 2);
  f16*   vpT16  = (f16*)alloc(24L * 64 * 256 * 2);
  float* x1     = (float*)alloc(16384L * 768 * 4);
  f16*   big16  = (f16*)alloc(16384L * 3072 * 2);  // qkv16 then g16 (reused)
  f16*   qkv16  = big16;
  f16*   g16    = big16;

  // ---- weight casts ----
  {
    long n;
    n = 768L * 2304;
    cast_f16_kernel<<<dim3((unsigned)((n + 255) / 256)), 256, 0, stream>>>(qkv_w, qkvW16, n);
    n = 768L * 768;
    cast_f16_kernel<<<dim3((unsigned)((n + 255) / 256)), 256, 0, stream>>>(proj_w, projW16, n);
    n = 768L * 3072;
    cast_f16_kernel<<<dim3((unsigned)((n + 255) / 256)), 256, 0, stream>>>(fc1_w, fc1W16, n);
    n = 3072L * 768;
    cast_f16_kernel<<<dim3((unsigned)((n + 255) / 256)), 256, 0, stream>>>(fc2_w, fc2W16, n);
    n = 8192L * 256;
    transpose_cast_kernel<<<dim3((unsigned)((n + 255) / 256)), 256, 0, stream>>>(Ek, EkT16, 8192, 256);
    transpose_cast_kernel<<<dim3((unsigned)((n + 255) / 256)), 256, 0, stream>>>(Ev, EvT16, 8192, 256);
  }

  // ---- attention branch ----
  ln_cast_kernel<<<16384, 256, 0, stream>>>(x, ln1_w, ln1_b, hbuf16);
  gemm_kernel<0><<<dim3(18, 128), 256, 0, stream>>>(hbuf16, 768, qkvW16, 2304, qkv_b,
                                                    nullptr, qkv16, nullptr, 2304, 768);
  gemm_kv_kernel<false><<<dim3(2, 24), 256, 0, stream>>>(EkT16, qkv16, 1, kp16);
  gemm_kv_kernel<true><<<dim3(2, 24), 256, 0, stream>>>(EvT16, qkv16, 2, vpT16);
  attn_kernel<<<dim3(64, 12, 2), 256, 131072, stream>>>(qkv16, kp16, vpT16, hbuf16);
  gemm_kernel<2><<<dim3(6, 128), 256, 0, stream>>>(hbuf16, 768, projW16, 768, proj_b, x,
                                                   nullptr, x1, 768, 768);

  // ---- MLP branch ----
  ln_cast_kernel<<<16384, 256, 0, stream>>>(x1, ln2_w, ln2_b, hbuf16);
  gemm_kernel<1><<<dim3(24, 128), 256, 0, stream>>>(hbuf16, 768, fc1W16, 3072, fc1_b,
                                                    nullptr, g16, nullptr, 3072, 768);
  gemm_kernel<2><<<dim3(6, 128), 256, 0, stream>>>(g16, 3072, fc2W16, 768, fc2_b, x1,
                                                   nullptr, (float*)d_out, 768, 3072);
}
